// Block_32710470926774
// MI455X (gfx1250) — matrix operand order repaired and checked by reading, not compiled, not run on hardware
//
#include <hip/hip_runtime.h>
#include <math.h>

// ---------------- constants ----------------
#define WSZ    14          // window size
#define NHEAD  12
#define HD     64          // head dim
#define CDIM   768
#define NTOK   196         // WSZ*WSZ
#define NTOKP  208         // padded to 13*16 (M tiling)
#define NTOKP2 224         // padded to 7*32  (K tiling of P@V)
#define NWIN   200         // 8 * 5 * 5
#define IMG    64
#define MROWS  (NWIN*NTOKP)   // 41600, /64 = 650 tiles
#define TOKENS (8*IMG*IMG)    // 32768, /64 = 512 tiles

// ---------------- WMMA types (CDNA5, wave32) ----------------
typedef __bf16 bf16t;
typedef bf16t v16bf __attribute__((ext_vector_type(16)));
typedef float v8f   __attribute__((ext_vector_type(8)));

union FragAB { v16bf v; uint4 q[2]; };

__device__ __forceinline__ float bf2f(unsigned short h) {
    union { unsigned int u; float f; } x; x.u = ((unsigned int)h) << 16; return x.f;
}
__device__ __forceinline__ unsigned short f2bf(float f) {
    union { float f; unsigned int u; } x; x.f = f;
    unsigned int u = x.u;
    unsigned int r = u + 0x7FFFu + ((u >> 16) & 1u);          // round-to-nearest-even
    if ((u & 0x7F800000u) == 0x7F800000u) r = u;              // inf/nan: truncate
    return (unsigned short)(r >> 16);
}
__device__ __forceinline__ v8f wmma_bf(FragAB a, FragAB b, v8f c) {
    return __builtin_amdgcn_wmma_f32_16x16x32_bf16(false, a.v, false, b.v,
                                                   (short)0, c, false, false);
}

// A fragment (16x32 bf16, M x K): lane m=lane&15, half=lane>>4.
// elements 0..7  -> K = kb + 8*half + 0..7   (16B load)
// elements 8..15 -> K = kb + 16 + 8*half + 0..7 (16B load)
__device__ __forceinline__ FragAB load_a(const unsigned short* row, int kb, int half) {
    FragAB f;
    const unsigned short* p = row + kb + 8 * half;
    f.q[0] = *(const uint4*)p;
    f.q[1] = *(const uint4*)(p + 16);
    return f;
}
// B fragment (32x16 bf16, K x N) from transposed weight Wt[N][K]:
// lane n=lane&15: elements 0..7 -> K = kb + 8*half + 0..7, elements 8..15 -> K = kb + 16 + 8*half + 0..7.
__device__ __forceinline__ FragAB load_b(const unsigned short* nrow, int kb, int half) {
    FragAB f;
    const unsigned short* p = nrow + kb + 8 * half;
    f.q[0] = *(const uint4*)p;
    f.q[1] = *(const uint4*)(p + 16);
    return f;
}

// 64x64 output tile per wave: 16 accumulators, 4 A-frags + 4 B-frags per k-step
// -> 16 WMMAs per 8 fragment loads (1.0 wmma per b128 load pair-half).
__device__ __forceinline__ void gemm_core64(const unsigned short* __restrict__ A,
                                            const unsigned short* __restrict__ Wt,
                                            int K, int mbase, int n64, v8f acc[16]) {
    int lane = threadIdx.x, ln = lane & 15, half = lane >> 4;
    v8f z = {0.f,0.f,0.f,0.f,0.f,0.f,0.f,0.f};
#pragma unroll
    for (int i = 0; i < 16; i++) acc[i] = z;
    const unsigned short* arow0 = A + (size_t)(mbase + ln) * K;
    for (int kb = 0; kb < K; kb += 32) {
        FragAB a[4], b[4];
#pragma unroll
        for (int mi = 0; mi < 4; mi++)
            a[mi] = load_a(arow0 + (size_t)mi * 16 * K, kb, half);
#pragma unroll
        for (int nj = 0; nj < 4; nj++)
            b[nj] = load_b(Wt + (size_t)(n64 * 64 + nj * 16 + ln) * K, kb, half);
#pragma unroll
        for (int mi = 0; mi < 4; mi++)
#pragma unroll
            for (int nj = 0; nj < 4; nj++)
                acc[mi * 4 + nj] = wmma_bf(a[mi], b[nj], acc[mi * 4 + nj]);
    }
}

// ---------------- weight transpose + f32->bf16 ----------------
__global__ void k_wt(const float* __restrict__ W, unsigned short* __restrict__ Wt,
                     int K, int N) {
    long long i = (long long)blockIdx.x * 256 + threadIdx.x;
    if (i >= (long long)K * N) return;
    int n = (int)(i % N), k = (int)(i / N);
    Wt[(size_t)n * K + k] = f2bf(W[i]);
}

__global__ void k_zero32(unsigned int* p, long long n) {
    long long i = (long long)blockIdx.x * 256 + threadIdx.x;
    if (i < n) p[i] = 0u;
}

// ---------------- LN1 -> bf16 windowed layout ----------------
__global__ __launch_bounds__(128) void k_ln1(const float* __restrict__ x,
                                             const float* __restrict__ g,
                                             const float* __restrict__ bvec,
                                             unsigned short* __restrict__ Xwin) {
    int row = blockIdx.x * 4 + (threadIdx.x >> 5);
    int lane = threadIdx.x & 31;
    if (row >= MROWS) return;
    int win = row / NTOKP, t = row % NTOKP;
    int b = win / 25, wi = (win % 25) / 5, wj = win % 5;
    int h = wi * WSZ + t / WSZ, w = wj * WSZ + t % WSZ;
    unsigned short* dst = Xwin + (size_t)row * CDIM;
    bool valid = (t < NTOK) && (h < IMG) && (w < IMG);
    if (!valid) {
        for (int i = lane; i < CDIM; i += 32) dst[i] = 0;
        return;
    }
    const float* src = x + ((((size_t)b * IMG) + h) * IMG + w) * CDIM;
    float vals[24], s = 0.f, ss = 0.f;
#pragma unroll
    for (int i = 0; i < 24; i++) { float v = src[lane + 32 * i]; vals[i] = v; s += v; ss += v * v; }
    for (int m = 16; m >= 1; m >>= 1) { s += __shfl_xor(s, m, 32); ss += __shfl_xor(ss, m, 32); }
    float mean = s * (1.f / CDIM);
    float var  = ss * (1.f / CDIM) - mean * mean;
    float rs = rsqrtf(var + 1e-5f);
#pragma unroll
    for (int i = 0; i < 24; i++) {
        int c = lane + 32 * i;
        dst[c] = f2bf((vals[i] - mean) * rs * g[c] + bvec[c]);
    }
}

// ---------------- LN2 (f32 rows -> bf16 rows) ----------------
__global__ __launch_bounds__(128) void k_ln2(const float* __restrict__ xr,
                                             const float* __restrict__ g,
                                             const float* __restrict__ bvec,
                                             unsigned short* __restrict__ X3) {
    int row = blockIdx.x * 4 + (threadIdx.x >> 5);
    int lane = threadIdx.x & 31;
    if (row >= TOKENS) return;
    const float* src = xr + (size_t)row * CDIM;
    unsigned short* dst = X3 + (size_t)row * CDIM;
    float vals[24], s = 0.f, ss = 0.f;
#pragma unroll
    for (int i = 0; i < 24; i++) { float v = src[lane + 32 * i]; vals[i] = v; s += v; ss += v * v; }
    for (int m = 16; m >= 1; m >>= 1) { s += __shfl_xor(s, m, 32); ss += __shfl_xor(ss, m, 32); }
    float mean = s * (1.f / CDIM);
    float var  = ss * (1.f / CDIM) - mean * mean;
    float rs = rsqrtf(var + 1e-5f);
#pragma unroll
    for (int i = 0; i < 24; i++) {
        int c = lane + 32 * i;
        dst[c] = f2bf((vals[i] - mean) * rs * g[c] + bvec[c]);
    }
}

// ---------------- QKV GEMM ----------------
// grid (650, 36): y = which*12 + head (qkv_w cols ordered [3][NH][hd]).
// Q/K stored row-major [wh][208][64]; V stored transposed [wh][64][224].
__global__ __launch_bounds__(32) void k_qkv(const unsigned short* __restrict__ Xwin,
                                            const unsigned short* __restrict__ Wt,
                                            const float* __restrict__ bias,
                                            unsigned short* __restrict__ Q,
                                            unsigned short* __restrict__ Kb,
                                            unsigned short* __restrict__ Vt) {
    int mt = blockIdx.x, j = blockIdx.y;
    v8f acc[16];
    gemm_core64(Xwin, Wt, CDIM, mt * 64, j, acc);
    int lane = threadIdx.x, ln = lane & 15, half = lane >> 4;
    int which = j / NHEAD, head = j % NHEAD;
#pragma unroll
    for (int mi = 0; mi < 4; mi++) {
        int mb = mt * 64 + mi * 16;              // 16-aligned => stays in one window
        int win = mb / NTOKP, t0 = mb % NTOKP;
        size_t wh = (size_t)win * NHEAD + head;
#pragma unroll
        for (int jj = 0; jj < 4; jj++) {
            int c = jj * 16 + ln;
            float bv = bias[j * 64 + c];
            const v8f av = acc[mi * 4 + jj];
            if (which == 2) {
                alignas(16) unsigned short tmp[8];
#pragma unroll
                for (int r = 0; r < 8; r++) tmp[r] = f2bf(av[r] + bv);
                unsigned short* dst = Vt + (wh * HD + c) * NTOKP2 + t0 + 8 * half;
                *(uint4*)dst = *(const uint4*)tmp;   // 8 consecutive token rows, contiguous
            } else {
                unsigned short* dst = ((which == 0) ? Q : Kb) + (wh * NTOKP + t0 + 8 * half) * HD + c;
#pragma unroll
                for (int r = 0; r < 8; r++) dst[(size_t)r * HD] = f2bf(av[r] + bv);
            }
        }
    }
}

// ---------------- attention: one wave per (win, head, 16-row q tile) ----------------
__global__ __launch_bounds__(32) void k_attn(const unsigned short* __restrict__ Q,
                                             const unsigned short* __restrict__ Kb,
                                             const unsigned short* __restrict__ Vt,
                                             const float* __restrict__ relH,
                                             const float* __restrict__ relW,
                                             unsigned short* __restrict__ X2) {
    __shared__ float biasH[16][WSZ];
    __shared__ float biasW[16][WSZ];
    __shared__ __align__(16) unsigned short P[16][NTOKP2];

    int bid = blockIdx.x;
    int mt = bid % 13, wh = bid / 13;
    int head = wh % NHEAD, win = wh / NHEAD;
    int lane = threadIdx.x, ln = lane & 15, half = lane >> 4;
    int mbase = mt * 16;

    const unsigned short* Qw = Q + (size_t)wh * NTOKP * HD;
    const unsigned short* Kw = Kb + (size_t)wh * NTOKP * HD;
    const unsigned short* Vw = Vt + (size_t)wh * HD * NTOKP2;

    // decomposed rel-pos bias for this q-row tile (uses UNscaled q, per reference)
    for (int idx = lane; idx < 16 * WSZ; idx += 32) {
        int lm = idx / WSZ, k = idx % WSZ;
        int qt = mbase + lm; if (qt > NTOK - 1) qt = NTOK - 1;   // clamp index math only
        int hq = qt / WSZ, wq = qt % WSZ;
        const unsigned short* qp = Qw + (size_t)(mbase + lm) * HD;
        const float* rh = relH + (size_t)(hq - k + WSZ - 1) * HD;
        const float* rw = relW + (size_t)(wq - k + WSZ - 1) * HD;
        float sh = 0.f, sw = 0.f;
        for (int c = 0; c < HD; c++) { float qv = bf2f(qp[c]); sh += qv * rh[c]; sw += qv * rw[c]; }
        biasH[lm][k] = sh; biasW[lm][k] = sw;
    }
    __syncthreads();

    // q A-fragments (K = 0..31, 32..63)
    const unsigned short* qrow = Qw + (size_t)(mbase + ln) * HD;
    FragAB a0 = load_a(qrow, 0, half);
    FragAB a1 = load_a(qrow, 32, half);

    // S = q @ k^T  (k row-major acts as transposed B operand)
    v8f acc[13];
    for (int nt = 0; nt < 13; nt++) {
        const unsigned short* krow = Kw + (size_t)(nt * 16 + ln) * HD;
        FragAB b0 = load_b(krow, 0, half);
        FragAB b1 = load_b(krow, 32, half);
        v8f z = {0.f,0.f,0.f,0.f,0.f,0.f,0.f,0.f};
        z = wmma_bf(a0, b0, z);
        acc[nt] = wmma_bf(a1, b1, z);
    }

    // scale + bias + key mask
    const float scale = 0.125f;   // hd^-0.5
    for (int nt = 0; nt < 13; nt++) {
        int kt = nt * 16 + ln;
        bool kv = kt < NTOK;
        int hk = kt / WSZ, wk = kt % WSZ;
#pragma unroll
        for (int r = 0; r < 8; r++) {
            int lm = 8 * half + r;
            float s = kv ? (scale * acc[nt][r] + biasH[lm][hk] + biasW[lm][wk]) : -1e30f;
            acc[nt][r] = s;
        }
    }

    // softmax: rows live in 16-lane halves of the C layout -> 4-step xor reductions
    float rowinv[8];
#pragma unroll
    for (int r = 0; r < 8; r++) {
        float m = -1e30f;
        for (int nt = 0; nt < 13; nt++) m = fmaxf(m, acc[nt][r]);
        m = fmaxf(m, __shfl_xor(m, 1, 32)); m = fmaxf(m, __shfl_xor(m, 2, 32));
        m = fmaxf(m, __shfl_xor(m, 4, 32)); m = fmaxf(m, __shfl_xor(m, 8, 32));
        float s = 0.f;
        for (int nt = 0; nt < 13; nt++) { float p = __expf(acc[nt][r] - m); acc[nt][r] = p; s += p; }
        s += __shfl_xor(s, 1, 32); s += __shfl_xor(s, 2, 32);
        s += __shfl_xor(s, 4, 32); s += __shfl_xor(s, 8, 32);
        rowinv[r] = 1.f / s;      // fold after P@V (linear per row)
    }

    // stage P to LDS as bf16 (C layout -> row-major), zero pad cols 208..223
    for (int nt = 0; nt < 13; nt++)
#pragma unroll
        for (int r = 0; r < 8; r++)
            P[8 * half + r][nt * 16 + ln] = f2bf(acc[nt][r]);
    for (int idx = lane; idx < 16 * 16; idx += 32)
        P[idx / 16][NTOKP + (idx % 16)] = 0;
    __syncthreads();

    // O = P @ V  (Vt rows are contiguous B fragments)
    v8f o[4];
    { v8f z = {0.f,0.f,0.f,0.f,0.f,0.f,0.f,0.f}; o[0]=z; o[1]=z; o[2]=z; o[3]=z; }
    for (int kt2 = 0; kt2 < 7; kt2++) {
        int kb = kt2 * 32;
        FragAB a;
        const unsigned short* pp = &P[ln][kb + 8 * half];
        a.q[0] = *(const uint4*)pp;
        a.q[1] = *(const uint4*)(pp + 16);
#pragma unroll
        for (int j = 0; j < 4; j++) {
            const unsigned short* pv = Vw + (size_t)(j * 16 + ln) * NTOKP2 + kb + 8 * half;
            FragAB b; b.q[0] = *(const uint4*)pv; b.q[1] = *(const uint4*)(pv + 16);
            o[j] = wmma_bf(a, b, o[j]);
        }
    }

    // store to X2[win][tok][head*64 + c] (valid tokens only)
    unsigned short* dst = X2 + (size_t)win * NTOKP * CDIM + head * HD;
#pragma unroll
    for (int j = 0; j < 4; j++) {
        int cc = j * 16 + ln;
#pragma unroll
        for (int r = 0; r < 8; r++) {
            int t = mbase + 8 * half + r;
            if (t < NTOK) dst[(size_t)t * CDIM + cc] = f2bf(o[j][r] * rowinv[r]);
        }
    }
}

// ---------------- proj GEMM + window-reverse + residual ----------------
__global__ __launch_bounds__(32) void k_proj(const unsigned short* __restrict__ X2,
                                             const unsigned short* __restrict__ Wt,
                                             const float* __restrict__ bias,
                                             const float* __restrict__ x_in,
                                             float* __restrict__ x_res) {
    int mt = blockIdx.x, n64 = blockIdx.y;
    v8f acc[16];
    gemm_core64(X2, Wt, CDIM, mt * 64, n64, acc);
    int lane = threadIdx.x, ln = lane & 15, half = lane >> 4;
#pragma unroll
    for (int mi = 0; mi < 4; mi++) {
        int mb = mt * 64 + mi * 16;
        int win = mb / NTOKP, t0 = mb % NTOKP;
        int b = win / 25, wi = (win % 25) / 5, wj = win % 5;
#pragma unroll
        for (int jj = 0; jj < 4; jj++) {
            int g = n64 * 64 + jj * 16 + ln;
            float bv = bias[g];
            const v8f av = acc[mi * 4 + jj];
#pragma unroll
            for (int r = 0; r < 8; r++) {
                int t = t0 + 8 * half + r;
                if (t < NTOK) {
                    int h = wi * WSZ + t / WSZ, w = wj * WSZ + t % WSZ;
                    if (h < IMG && w < IMG) {
                        size_t idx = ((((size_t)b * IMG) + h) * IMG + w) * CDIM + g;
                        x_res[idx] = x_in[idx] + av[r] + bv;
                    }
                }
            }
        }
    }
}

// ---------------- MLP1 GEMM + exact GELU -> bf16 ----------------
__global__ __launch_bounds__(32) void k_mlp1(const unsigned short* __restrict__ X3,
                                             const unsigned short* __restrict__ Wt,
                                             const float* __restrict__ bias,
                                             unsigned short* __restrict__ H1) {
    int mt = blockIdx.x, n64 = blockIdx.y;
    v8f acc[16];
    gemm_core64(X3, Wt, CDIM, mt * 64, n64, acc);
    int lane = threadIdx.x, ln = lane & 15, half = lane >> 4;
#pragma unroll
    for (int mi = 0; mi < 4; mi++) {
        int mb = mt * 64 + mi * 16;
#pragma unroll
        for (int jj = 0; jj < 4; jj++) {
            int g = n64 * 64 + jj * 16 + ln;
            float bv = bias[g];
            const v8f av = acc[mi * 4 + jj];
#pragma unroll
            for (int r = 0; r < 8; r++) {
                float v = av[r] + bv;
                float gl = 0.5f * v * (1.f + erff(v * 0.70710678118654752f));
                H1[(size_t)(mb + 8 * half + r) * (4 * CDIM) + g] = f2bf(gl);
            }
        }
    }
}

// ---------------- MLP2 GEMM + residual -> f32 output ----------------
__global__ __launch_bounds__(32) void k_mlp2(const unsigned short* __restrict__ H1,
                                             const unsigned short* __restrict__ Wt,
                                             const float* __restrict__ bias,
                                             const float* __restrict__ x_res,
                                             float* __restrict__ out) {
    int mt = blockIdx.x, n64 = blockIdx.y;
    v8f acc[16];
    gemm_core64(H1, Wt, 4 * CDIM, mt * 64, n64, acc);
    int lane = threadIdx.x, ln = lane & 15, half = lane >> 4;
#pragma unroll
    for (int mi = 0; mi < 4; mi++) {
        int mb = mt * 64 + mi * 16;
#pragma unroll
        for (int jj = 0; jj < 4; jj++) {
            int g = n64 * 64 + jj * 16 + ln;
            float bv = bias[g];
            const v8f av = acc[mi * 4 + jj];
#pragma unroll
            for (int r = 0; r < 8; r++) {
                size_t idx = (size_t)(mb + 8 * half + r) * CDIM + g;
                out[idx] = x_res[idx] + av[r] + bv;
            }
        }
    }
}

// ---------------- host launch ----------------
extern "C" void kernel_launch(void* const* d_in, const int* in_sizes, int n_in,
                              void* d_out, int out_size, void* d_ws, size_t ws_size,
                              hipStream_t stream) {
    const float* x      = (const float*)d_in[0];
    const float* n1g    = (const float*)d_in[1];
    const float* n1b    = (const float*)d_in[2];
    const float* qkv_w  = (const float*)d_in[3];
    const float* qkv_b  = (const float*)d_in[4];
    const float* proj_w = (const float*)d_in[5];
    const float* proj_b = (const float*)d_in[6];
    const float* relh   = (const float*)d_in[7];
    const float* relw   = (const float*)d_in[8];
    const float* n2g    = (const float*)d_in[9];
    const float* n2b    = (const float*)d_in[10];
    const float* w1     = (const float*)d_in[11];
    const float* b1     = (const float*)d_in[12];
    const float* w2     = (const float*)d_in[13];
    const float* b2     = (const float*)d_in[14];
    float* out = (float*)d_out;

    char* ws = (char*)d_ws;
    size_t off = 0;
    auto alloc = [&](size_t bytes) -> char* {
        char* p = ws + off;
        off = (off + bytes + 255) & ~(size_t)255;
        return p;
    };
    unsigned short* Xwin  = (unsigned short*)alloc((size_t)MROWS * CDIM * 2);
    unsigned short* qkvWt = (unsigned short*)alloc((size_t)3 * CDIM * CDIM * 2);
    unsigned short* projWt= (unsigned short*)alloc((size_t)CDIM * CDIM * 2);
    unsigned short* w1t   = (unsigned short*)alloc((size_t)4 * CDIM * CDIM * 2);
    unsigned short* w2t   = (unsigned short*)alloc((size_t)4 * CDIM * CDIM * 2);
    unsigned short* Qb    = (unsigned short*)alloc((size_t)NWIN * NHEAD * NTOKP * HD * 2);
    unsigned short* Kbuf  = (unsigned short*)alloc((size_t)NWIN * NHEAD * NTOKP * HD * 2);
    unsigned short* Vt    = (unsigned short*)alloc((size_t)NWIN * NHEAD * HD * NTOKP2 * 2);
    unsigned short* X2    = (unsigned short*)alloc((size_t)MROWS * CDIM * 2);
    float*          xres  = (float*)alloc((size_t)TOKENS * CDIM * 4);
    unsigned short* X3    = (unsigned short*)alloc((size_t)TOKENS * CDIM * 2);
    unsigned short* H1    = (unsigned short*)alloc((size_t)TOKENS * 4 * CDIM * 2);

    // 1) weights -> bf16 transposed
    {
        long long n;
        n = (long long)CDIM * (3 * CDIM);
        k_wt<<<dim3((unsigned)((n + 255) / 256)), 256, 0, stream>>>(qkv_w, qkvWt, CDIM, 3 * CDIM);
        n = (long long)CDIM * CDIM;
        k_wt<<<dim3((unsigned)((n + 255) / 256)), 256, 0, stream>>>(proj_w, projWt, CDIM, CDIM);
        n = (long long)CDIM * (4 * CDIM);
        k_wt<<<dim3((unsigned)((n + 255) / 256)), 256, 0, stream>>>(w1, w1t, CDIM, 4 * CDIM);
        n = (long long)(4 * CDIM) * CDIM;
        k_wt<<<dim3((unsigned)((n + 255) / 256)), 256, 0, stream>>>(w2, w2t, 4 * CDIM, CDIM);
    }
    // 2) zero Vt (pad columns 208..223 must be zero for P@V)
    {
        long long n32 = (long long)NWIN * NHEAD * HD * NTOKP2 * 2 / 4;
        k_zero32<<<dim3((unsigned)((n32 + 255) / 256)), 256, 0, stream>>>((unsigned int*)Vt, n32);
    }
    // 3) LN1 -> windowed bf16 (padded rows zeroed)
    k_ln1<<<dim3(MROWS / 4), 128, 0, stream>>>(x, n1g, n1b, Xwin);
    // 4) QKV GEMM (64x64 tiles)
    k_qkv<<<dim3(MROWS / 64, 36), 32, 0, stream>>>(Xwin, qkvWt, qkv_b, Qb, Kbuf, Vt);
    // 5) windowed attention (2400 win-heads x 13 q tiles)
    k_attn<<<dim3(NWIN * NHEAD * 13), 32, 0, stream>>>(Qb, Kbuf, Vt, relh, relw, X2);
    // 6) proj + window-reverse + residual
    k_proj<<<dim3(MROWS / 64, CDIM / 64), 32, 0, stream>>>(X2, projWt, proj_b, x, xres);
    // 7) LN2
    k_ln2<<<dim3(TOKENS / 4), 128, 0, stream>>>(xres, n2g, n2b, X3);
    // 8) MLP fc1 + GELU
    k_mlp1<<<dim3(TOKENS / 64, (4 * CDIM) / 64), 32, 0, stream>>>(X3, w1t, b1, H1);
    // 9) MLP fc2 + residual -> out
    k_mlp2<<<dim3(TOKENS / 64, CDIM / 64), 32, 0, stream>>>(H1, w2t, b2, xres, out);

    (void)in_sizes; (void)n_in; (void)out_size; (void)ws_size;
}
